// DeepMarkovModel_25451976196390
// MI455X (gfx1250) — matrix operand order repaired and checked by reading, not compiled, not run on hardware
//
#include <hip/hip_runtime.h>
#include <hip/hip_bf16.h>
#include <math.h>

// ---------------- constants for this problem ----------------
#define BB 512
#define TT 200
#define DD 128
#define LL 128
#define HH 512
#define LOG2PI_F 1.8378770664093453f
#define FREE_BITS 12.8f   // 0.1 * L

// ---------------- bf16 helpers (no scalar __bf16 arithmetic) ----------------
typedef __attribute__((ext_vector_type(16))) __bf16         v16bf;
typedef __attribute__((ext_vector_type(8)))  float          v8f;
typedef __attribute__((ext_vector_type(8)))  unsigned short v8u;
typedef __attribute__((ext_vector_type(16))) unsigned short v16u;

__device__ __forceinline__ unsigned short f2bf(float f) {
    union { float f; unsigned u; } v; v.f = f;
    unsigned u = v.u;
    unsigned r = u + 0x7FFFu + ((u >> 16) & 1u);   // round-to-nearest-even
    return (unsigned short)(r >> 16);
}
__device__ __forceinline__ float sigf(float x) { return 1.0f / (1.0f + expf(-x)); }

// Fragment loaders using contiguous k order (k = 16*half + s), the same for A and B.
// A 16x32 bf16: lanes 0-15 -> K {0..15}; lanes 16-31 -> K {16..31}
__device__ __forceinline__ v16bf load_a_frag(const unsigned short* __restrict__ arow,
                                             int k0, int half) {
    v16u au = *(const v16u*)(arow + k0 + half * 16);
    return __builtin_bit_cast(v16bf, au);
}
// B 32x16 bf16: lane group selects K half; 16 contiguous K elems from row-major W[N,K]
__device__ __forceinline__ v16bf load_b_frag(const unsigned short* __restrict__ wrow,
                                             int k0, int half) {
    v16u bu = *(const v16u*)(wrow + k0 + half * 16);
    return __builtin_bit_cast(v16bf, bu);
}

// ---------------- one-time conversion kernels ----------------
__global__ void k_cvt_f32_bf16(const float* __restrict__ s, unsigned short* __restrict__ d, int n) {
    int i = blockIdx.x * blockDim.x + threadIdx.x;
    if (i < n) d[i] = f2bf(s[i]);
}

// Wcat[n, 0:128] = Wih[n, :], Wcat[n, 128:640] = Whh[n, :]   (n = 0..2047)
__global__ void k_build_wcat(const float* __restrict__ wih, const float* __restrict__ whh,
                             unsigned short* __restrict__ w) {
    int idx = blockIdx.x * blockDim.x + threadIdx.x;   // 2048*640
    if (idx >= 2048 * 640) return;
    int n = idx / 640, k = idx - n * 640;
    float v = (k < DD) ? wih[n * DD + k] : whh[n * HH + (k - DD)];
    w[idx] = f2bf(v);
}

// xall[(t*B + b)*D + d] = bf16( x_seq[b, t, d] )
__global__ void k_x_transpose(const float* __restrict__ x, unsigned short* __restrict__ d, int total) {
    int idx = blockIdx.x * blockDim.x + threadIdx.x;
    if (idx >= total) return;
    int dd = idx % DD;
    int r  = idx / DD;
    int b  = r % BB;
    int t  = r / BB;
    d[idx] = f2bf(x[((size_t)b * TT + t) * DD + dd]);
}

// copy x_t into the x-region of the [x|h] concat activation buffer (row stride 640)
__global__ void k_copy_x_step(const unsigned short* __restrict__ src, unsigned short* __restrict__ xhcat) {
    int idx = blockIdx.x * blockDim.x + threadIdx.x;   // B*D
    int b = idx >> 7, d = idx & 127;
    xhcat[b * 640 + d] = src[idx];
}

// ---------------- WMMA GEMM:  C[M,N] = act( A[M,K] @ W[N,K]^T + bias[N] ) ----------------
// 32x32 register blocking per wave (2x2 WMMA tiles -> 4 WMMAs per 8 b128 loads),
// 8 waves per block, K-loop step 32.  M,N multiples of 32; K multiple of 32.
template <int RELU, int OUTBF>
__global__ __launch_bounds__(256) void k_gemm_wmma(
    const unsigned short* __restrict__ A, int lda,
    const unsigned short* __restrict__ W, int ldw,
    const float* __restrict__ bias,
    void* __restrict__ Cout, int ldc,
    int M, int N, int K)
{
    const int wave = threadIdx.x >> 5;
    const int lane = threadIdx.x & 31;
    const int half = lane >> 4;          // 0 or 1
    const int l16  = lane & 15;

    const int tilesN = N >> 5;                 // 32-wide macro tiles along N
    const int total  = (M >> 5) * tilesN;
    int tile = blockIdx.x * 8 + wave;
    if (tile >= total) return;                 // wave-uniform exit -> EXEC all-1s for WMMA
    const int tM = tile / tilesN;
    const int tN = tile - tM * tilesN;

    const unsigned short* arow0 = A + (size_t)(tM * 32 + l16) * lda;
    const unsigned short* arow1 = arow0 + (size_t)16 * lda;
    const unsigned short* wrow0 = W + (size_t)(tN * 32 + l16) * ldw;
    const unsigned short* wrow1 = wrow0 + (size_t)16 * ldw;

    v8f acc00 = {}, acc01 = {}, acc10 = {}, acc11 = {};
    for (int k0 = 0; k0 < K; k0 += 32) {
        v16bf a0 = load_a_frag(arow0, k0, half);
        v16bf a1 = load_a_frag(arow1, k0, half);
        v16bf b0 = load_b_frag(wrow0, k0, half);
        v16bf b1 = load_b_frag(wrow1, k0, half);
        acc00 = __builtin_amdgcn_wmma_f32_16x16x32_bf16(false, a0, false, b0, (short)0, acc00, false, false);
        acc01 = __builtin_amdgcn_wmma_f32_16x16x32_bf16(false, a0, false, b1, (short)0, acc01, false, false);
        acc10 = __builtin_amdgcn_wmma_f32_16x16x32_bf16(false, a1, false, b0, (short)0, acc10, false, false);
        acc11 = __builtin_amdgcn_wmma_f32_16x16x32_bf16(false, a1, false, b1, (short)0, acc11, false, false);
    }

    const int n0 = tN * 32 + l16;
    const int n1 = n0 + 16;
    const float bb0 = bias ? bias[n0] : 0.0f;
    const float bb1 = bias ? bias[n1] : 0.0f;
#pragma unroll
    for (int v = 0; v < 8; ++v) {
        int m0 = tM * 32 + half * 8 + v;       // C/D layout: lanes 16-31 hold M+8
        int m1 = m0 + 16;
        float v00 = acc00[v] + bb0, v01 = acc01[v] + bb1;
        float v10 = acc10[v] + bb0, v11 = acc11[v] + bb1;
        if (RELU) {
            v00 = fmaxf(v00, 0.0f); v01 = fmaxf(v01, 0.0f);
            v10 = fmaxf(v10, 0.0f); v11 = fmaxf(v11, 0.0f);
        }
        if (OUTBF) {
            unsigned short* C = (unsigned short*)Cout;
            C[(size_t)m0 * ldc + n0] = f2bf(v00);
            C[(size_t)m0 * ldc + n1] = f2bf(v01);
            C[(size_t)m1 * ldc + n0] = f2bf(v10);
            C[(size_t)m1 * ldc + n1] = f2bf(v11);
        } else {
            float* C = (float*)Cout;
            C[(size_t)m0 * ldc + n0] = v00;
            C[(size_t)m0 * ldc + n1] = v01;
            C[(size_t)m1 * ldc + n0] = v10;
            C[(size_t)m1 * ldc + n1] = v11;
        }
    }
}

template <int RELU, int OUTBF>
static inline void gemm(hipStream_t s,
                        const unsigned short* A, int lda,
                        const unsigned short* W, int ldw,
                        const float* bias, void* C, int ldc,
                        int M, int N, int K)
{
    int tiles  = (M / 32) * (N / 32);
    int blocks = (tiles + 7) / 8;
    k_gemm_wmma<RELU, OUTBF><<<blocks, 256, 0, s>>>(A, lda, W, ldw, bias, C, ldc, M, N, K);
}

// ---------------- LSTM pointwise: gates -> (h, c); h written into both concat buffers ----------------
__global__ void k_lstm(const float* __restrict__ gates, const float* __restrict__ bl,
                       float* __restrict__ c,
                       unsigned short* __restrict__ xhcat, unsigned short* __restrict__ zcat)
{
    int idx = blockIdx.x * blockDim.x + threadIdx.x;   // B*H
    int b = idx >> 9, j = idx & 511;
    const float gi = gates[b * 2048 + j]          + bl[j];
    const float gf = gates[b * 2048 + 512 + j]    + bl[512 + j];
    const float gg = gates[b * 2048 + 1024 + j]   + bl[1024 + j];
    const float go = gates[b * 2048 + 1536 + j]   + bl[1536 + j];
    float cn = sigf(gf) * c[idx] + sigf(gi) * tanhf(gg);
    c[idx] = cn;
    float h = sigf(go) * tanhf(cn);
    unsigned short hb = f2bf(h);
    xhcat[b * 640 + 128 + j] = hb;
    zcat [b * 640 + 128 + j] = hb;
}

// ---------------- reparameterize + KL accumulation (one block per batch row) ----------------
__global__ void k_reparam_kl(const float* __restrict__ q, const float* __restrict__ p,
                             const float* __restrict__ z1loc, const float* __restrict__ z1ls,
                             const float* __restrict__ eps_t,
                             unsigned short* __restrict__ zbuf, unsigned short* __restrict__ zcat,
                             float* __restrict__ kl, int isT0)
{
    __shared__ float sred[LL];
    const int b = blockIdx.x, l = threadIdx.x;         // L = 128 threads
    const float qloc = q[b * 256 + l];
    const float qls  = q[b * 256 + 128 + l];
    const float z    = qloc + expf(qls) * eps_t[b * LL + l];
    unsigned short zb = f2bf(z);
    zbuf[b * LL + l]  = zb;
    zcat[b * 640 + l] = zb;

    float ploc, pls;
    if (isT0) { ploc = z1loc[l]; pls = z1ls[l]; }
    else      { ploc = p[b * 256 + l]; pls = p[b * 256 + 128 + l]; }
    const float dq = (z - qloc) * expf(-qls);
    const float dp = (z - ploc) * expf(-pls);
    sred[l] = -0.5f * (dq * dq + 2.0f * qls) + 0.5f * (dp * dp + 2.0f * pls);  // LOG2PI cancels
    __syncthreads();
    for (int s = 64; s > 0; s >>= 1) { if (l < s) sred[l] += sred[l + s]; __syncthreads(); }
    if (l == 0) {
        float contrib = fmaxf(sred[0], FREE_BITS);
        if (isT0) kl[b] = contrib; else kl[b] += contrib;
    }
}

// ---------------- emission log-prob accumulation ----------------
__global__ void k_recon(const float* __restrict__ e, const float* __restrict__ x_seq,
                        float* __restrict__ recon, int t, int isT0)
{
    __shared__ float sred[DD];
    const int b = blockIdx.x, d = threadIdx.x;         // D = 128 threads
    const float eloc = e[b * 256 + d];
    const float els  = e[b * 256 + 128 + d];
    const float x    = x_seq[((size_t)b * TT + t) * DD + d];
    const float dv   = (x - eloc) * expf(-els);
    sred[d] = -0.5f * (dv * dv + 2.0f * els + LOG2PI_F);
    __syncthreads();
    for (int s = 64; s > 0; s >>= 1) { if (d < s) sred[d] += sred[d + s]; __syncthreads(); }
    if (d == 0) { if (isT0) recon[b] = sred[0]; else recon[b] += sred[0]; }
}

// ---------------- final reduction: mean(kl - recon) ----------------
__global__ void k_final(const float* __restrict__ kl, const float* __restrict__ recon,
                        float* __restrict__ out)
{
    __shared__ float sred[BB];
    int i = threadIdx.x;                                // 512 threads
    sred[i] = kl[i] - recon[i];
    __syncthreads();
    for (int s = 256; s > 0; s >>= 1) { if (i < s) sred[i] += sred[i + s]; __syncthreads(); }
    if (i == 0) out[0] = sred[0] / (float)BB;
}

// ---------------- host orchestration ----------------
extern "C" void kernel_launch(void* const* d_in, const int* in_sizes, int n_in,
                              void* d_out, int out_size, void* d_ws, size_t ws_size,
                              hipStream_t stream)
{
    const float* x_seq = (const float*)d_in[0];
    const float* eps   = (const float*)d_in[1];
    const float* Wih   = (const float*)d_in[2];
    const float* Whh   = (const float*)d_in[3];
    const float* b_lstm= (const float*)d_in[4];
    const float* Wc0 = (const float*)d_in[5];  const float* bc0 = (const float*)d_in[6];
    const float* Wc1 = (const float*)d_in[7];  const float* bc1 = (const float*)d_in[8];
    const float* Wc2 = (const float*)d_in[9];  const float* bc2 = (const float*)d_in[10];
    const float* Wt0 = (const float*)d_in[11]; const float* bt0 = (const float*)d_in[12];
    const float* Wt1 = (const float*)d_in[13]; const float* bt1 = (const float*)d_in[14];
    const float* Wt2 = (const float*)d_in[15]; const float* bt2 = (const float*)d_in[16];
    const float* We0 = (const float*)d_in[17]; const float* be0 = (const float*)d_in[18];
    const float* We1 = (const float*)d_in[19]; const float* be1 = (const float*)d_in[20];
    const float* We2 = (const float*)d_in[21]; const float* be2 = (const float*)d_in[22];
    const float* z1_loc = (const float*)d_in[23];
    const float* z1_ls  = (const float*)d_in[24];

    // bump allocator over workspace (256-byte aligned)
    char* base = (char*)d_ws;
    size_t off = 0;
    auto alloc = [&](size_t bytes) -> void* {
        void* p = base + off;
        off = (off + bytes + 255) & ~(size_t)255;
        return p;
    };

    unsigned short* wcat = (unsigned short*)alloc((size_t)2048 * 640 * 2);
    unsigned short* wc0b = (unsigned short*)alloc((size_t)HH * 640 * 2);
    unsigned short* wc1b = (unsigned short*)alloc((size_t)HH * HH * 2);
    unsigned short* wc2b = (unsigned short*)alloc((size_t)256 * HH * 2);
    unsigned short* wt0b = (unsigned short*)alloc((size_t)HH * LL * 2);
    unsigned short* wt1b = (unsigned short*)alloc((size_t)HH * HH * 2);
    unsigned short* wt2b = (unsigned short*)alloc((size_t)256 * HH * 2);
    unsigned short* we0b = (unsigned short*)alloc((size_t)HH * LL * 2);
    unsigned short* we1b = (unsigned short*)alloc((size_t)HH * HH * 2);
    unsigned short* we2b = (unsigned short*)alloc((size_t)256 * HH * 2);
    unsigned short* xall = (unsigned short*)alloc((size_t)TT * BB * DD * 2);
    unsigned short* xhcat= (unsigned short*)alloc((size_t)BB * 640 * 2);
    unsigned short* zcat = (unsigned short*)alloc((size_t)BB * 640 * 2);
    unsigned short* zbuf = (unsigned short*)alloc((size_t)BB * LL * 2);
    unsigned short* m1   = (unsigned short*)alloc((size_t)BB * HH * 2);
    unsigned short* m2   = (unsigned short*)alloc((size_t)BB * HH * 2);
    float* cstate = (float*)alloc((size_t)BB * HH * 4);
    float* gates  = (float*)alloc((size_t)BB * 2048 * 4);
    float* qbuf   = (float*)alloc((size_t)BB * 256 * 4);
    float* pbuf   = (float*)alloc((size_t)BB * 256 * 4);
    float* ebuf   = (float*)alloc((size_t)BB * 256 * 4);
    float* klb    = (float*)alloc((size_t)BB * 4);
    float* reconb = (float*)alloc((size_t)BB * 4);
    (void)ws_size; (void)in_sizes; (void)n_in; (void)out_size;

    // ---- one-time weight / input conversion (bf16) ----
    k_build_wcat<<<(2048 * 640 + 255) / 256, 256, 0, stream>>>(Wih, Whh, wcat);
    auto cvt = [&](const float* s, unsigned short* d, int n) {
        k_cvt_f32_bf16<<<(n + 255) / 256, 256, 0, stream>>>(s, d, n);
    };
    cvt(Wc0, wc0b, HH * 640); cvt(Wc1, wc1b, HH * HH); cvt(Wc2, wc2b, 256 * HH);
    cvt(Wt0, wt0b, HH * LL);  cvt(Wt1, wt1b, HH * HH); cvt(Wt2, wt2b, 256 * HH);
    cvt(We0, we0b, HH * LL);  cvt(We1, we1b, HH * HH); cvt(We2, we2b, 256 * HH);
    {
        int total = TT * BB * DD;
        k_x_transpose<<<(total + 255) / 256, 256, 0, stream>>>(x_seq, xall, total);
    }
    hipMemsetAsync(cstate, 0, (size_t)BB * HH * 4, stream);     // c0 = 0
    hipMemsetAsync(xhcat,  0, (size_t)BB * 640 * 2, stream);    // h0 = 0 (x-region overwritten)
    hipMemsetAsync(zcat,   0, (size_t)BB * 640 * 2, stream);    // z0 = 0 (h-region overwritten)

    // ---- sequential scan over time ----
    for (int t = 0; t < TT; ++t) {
        // stage x_t into the [x|h] concat activation
        k_copy_x_step<<<(BB * DD) / 256, 256, 0, stream>>>(xall + (size_t)t * BB * DD, xhcat);

        // gates = [x_t | h_{t-1}] @ [Wih | Whh]^T      (512 x 2048 x 640)
        gemm<0, 0>(stream, xhcat, 640, wcat, 640, nullptr, gates, 2048, BB, 2048, 640);
        k_lstm<<<(BB * HH) / 256, 256, 0, stream>>>(gates, b_lstm, cstate, xhcat, zcat);

        // combiner([z_{t-1} | h_t]): 640 -> 512 -> 512 -> 256
        gemm<1, 1>(stream, zcat, 640, wc0b, 640, bc0, m1, HH, BB, HH, 640);
        gemm<1, 1>(stream, m1, HH, wc1b, HH, bc1, m2, HH, BB, HH, HH);
        gemm<0, 0>(stream, m2, HH, wc2b, HH, bc2, qbuf, 256, BB, 256, HH);

        // transition(z_{t-1}) for t >= 1: 128 -> 512 -> 512 -> 256
        if (t > 0) {
            gemm<1, 1>(stream, zbuf, LL, wt0b, LL, bt0, m1, HH, BB, HH, LL);
            gemm<1, 1>(stream, m1, HH, wt1b, HH, bt1, m2, HH, BB, HH, HH);
            gemm<0, 0>(stream, m2, HH, wt2b, HH, bt2, pbuf, 256, BB, 256, HH);
        }

        // z_t = q_loc + exp(q_ls)*eps_t ; kl += max(lq - lp, free_bits)
        k_reparam_kl<<<BB, LL, 0, stream>>>(qbuf, pbuf, z1_loc, z1_ls,
                                            eps + (size_t)t * BB * LL,
                                            zbuf, zcat, klb, t == 0 ? 1 : 0);

        // emission(z_t): 128 -> 512 -> 512 -> 256 ; recon += gauss_lp(x_t, ...)
        gemm<1, 1>(stream, zbuf, LL, we0b, LL, be0, m1, HH, BB, HH, LL);
        gemm<1, 1>(stream, m1, HH, we1b, HH, be1, m2, HH, BB, HH, HH);
        gemm<0, 0>(stream, m2, HH, we2b, HH, be2, ebuf, 256, BB, 256, HH);
        k_recon<<<BB, DD, 0, stream>>>(ebuf, x_seq, reconb, t, t == 0 ? 1 : 0);
    }

    k_final<<<1, BB, 0, stream>>>(klb, reconb, (float*)d_out);
}
